// MultiheadAttention_63642825392189
// MI455X (gfx1250) — matrix operand order repaired and checked by reading, not compiled, not run on hardware
//
#include <hip/hip_runtime.h>
#include <hip/hip_bf16.h>
#include <stdint.h>

// ---------------------------------------------------------------------------
// CDNA5 (gfx1250) bf16 WMMA multi-head attention pipeline.
// Workspace layout (bytes):
//   e  : 3 * 49152 * 768 bf16  (embed output, per head)   = 226,492,416
//   o  : 49152 * 768 bf16      (attention output, concat) =  75,497,472
// Uses: v_wmma_f32_16x16x32_bf16, global_load_async_to_lds_b128 (ASYNCcnt).
// ---------------------------------------------------------------------------

typedef __attribute__((ext_vector_type(16))) __bf16        v16bf;
typedef __attribute__((ext_vector_type(8)))  float         v8f;
typedef __attribute__((ext_vector_type(8)))  unsigned short u16x8;
typedef __attribute__((ext_vector_type(16))) unsigned short u16x16;

#define DEVINL __device__ __forceinline__

DEVINL unsigned short f2bf(float f) {
  union { float f; unsigned int u; } c; c.f = f;
  unsigned int u = c.u;
  u += 0x7FFFu + ((u >> 16) & 1u);   // round-to-nearest-even
  return (unsigned short)(u >> 16);
}

DEVINL u16x16 concat8(u16x8 a, u16x8 b) {
  return __builtin_shufflevector(a, b, 0, 1, 2, 3, 4, 5, 6, 7,
                                 8, 9, 10, 11, 12, 13, 14, 15);
}

// A fragment (16x32 bf16, MxK): lane<16 -> K{0..7,16..23}, lane>=16 -> K{8..15,24..31}
DEVINL u16x16 load_a_frag(const unsigned short* rowp, int k0, int hi) {
  u16x8 a0 = *(const u16x8*)(rowp + k0 + hi * 8);
  u16x8 a1 = *(const u16x8*)(rowp + k0 + 16 + hi * 8);
  return concat8(a0, a1);
}

// B fragment (32x16 bf16, KxN): lane holds column n=lane&15; lane<16 -> K{0..7,16..23},
// lane>=16 -> K{8..15,24..31}. colp = this lane's column base (contiguous K).
DEVINL u16x16 load_b_frag(const unsigned short* colp, int k0, int hi) {
  u16x8 b0 = *(const u16x8*)(colp + k0 + hi * 8);
  u16x8 b1 = *(const u16x8*)(colp + k0 + 16 + hi * 8);
  return concat8(b0, b1);
}

DEVINL v8f wmma_bf16(u16x16 a, u16x16 b, v8f c) {
  return __builtin_amdgcn_wmma_f32_16x16x32_bf16(
      false, __builtin_bit_cast(v16bf, a),
      false, __builtin_bit_cast(v16bf, b),
      (short)0, c, false, false);
}

// Async global->LDS copy, 16 bytes per lane (GVS mode: SGPR64 base +
// 32-bit VGPR byte offset; LDS address taken from the first VGPR operand). Tracked by ASYNCcnt.
DEVINL void async_ld_b128(unsigned lds_off, unsigned goff, const void* base) {
  asm volatile("global_load_async_to_lds_b128 %0, %1, %2"
               :: "v"(lds_off), "v"(goff), "s"(base)
               : "memory");
}
DEVINL void wait_async0() {
  asm volatile("s_wait_asynccnt 0" ::: "memory");
}
DEVINL unsigned lds_off_of(const void* p) {  // addrspace(3) ptr -> byte offset
  return (unsigned)(uintptr_t)p;
}

// ---------------------------------------------------------------------------
// Kernel A: e[h, row, d] = sum_c x[row, c] * W_embed[h, d, c] + b_embed[h, d]
// rows = B*M = 49152, K = 256, N = 768 per head. 128x128 tiles, bf16 WMMA.
// (f32 -> bf16 conversion required, so staging stays on the VALU path.)
// ---------------------------------------------------------------------------
__global__ __launch_bounds__(256)
void mha_embed_gemm(const float* __restrict__ x,
                    const float* __restrict__ W_embed,
                    const float* __restrict__ b_embed,
                    unsigned short* __restrict__ e) {
  constexpr int LD = 264;  // 256 + 8 shorts -> 16B-aligned rows
  __shared__ __attribute__((aligned(16))) unsigned short As[128 * LD];
  __shared__ __attribute__((aligned(16))) unsigned short Bs[128 * LD];

  const int t = threadIdx.x;
  const int w = t >> 5, lane = t & 31, ln = lane & 15, hi = lane >> 4;
  const int rowBase = blockIdx.x * 128;
  const int dBase   = blockIdx.y * 128;
  const int h       = blockIdx.z;

  const float4* xg = (const float4*)x + (size_t)rowBase * 64;               // 64 f4/row
  const float4* wg = (const float4*)W_embed + ((size_t)h * 768 + dBase) * 64;
  for (int j = 0; j < 32; ++j) {
    int idx = t + j * 256;               // 0..8191
    int r = idx >> 6, c4 = idx & 63;
    float4 va = xg[(size_t)r * 64 + c4];
    unsigned short* pa = &As[r * LD + c4 * 4];
    pa[0] = f2bf(va.x); pa[1] = f2bf(va.y); pa[2] = f2bf(va.z); pa[3] = f2bf(va.w);
    float4 vb = wg[(size_t)r * 64 + c4];
    unsigned short* pb = &Bs[r * LD + c4 * 4];
    pb[0] = f2bf(vb.x); pb[1] = f2bf(vb.y); pb[2] = f2bf(vb.z); pb[3] = f2bf(vb.w);
  }
  __syncthreads();

  v8f acc[8] = {};
  const unsigned short* arow = &As[(w * 16 + ln) * LD];
#pragma unroll
  for (int kt = 0; kt < 8; ++kt) {
    u16x16 af = load_a_frag(arow, kt * 32, hi);
#pragma unroll
    for (int ct = 0; ct < 8; ++ct) {
      u16x16 bf = load_b_frag(&Bs[(ct * 16 + ln) * LD], kt * 32, hi);
      acc[ct] = wmma_bf16(af, bf, acc[ct]);
    }
  }

  const size_t hbase = (size_t)h * (49152ull * 768ull);
#pragma unroll
  for (int ct = 0; ct < 8; ++ct) {
#pragma unroll
    for (int r = 0; r < 8; ++r) {
      int row = w * 16 + r + 8 * hi;
      int col = ct * 16 + ln;
      float v = acc[ct][r] + b_embed[h * 768 + dBase + col];
      e[hbase + (size_t)(rowBase + row) * 768 + (dBase + col)] = f2bf(v);
    }
  }
}

// ---------------------------------------------------------------------------
// Kernel B: flash attention per (head, batch, 32-row q block).
// q/k/v are the 3 sequence chunks of e. SCALE = 256^-0.5 = 0.0625.
// Wave w: S tile (rt=w&1, keycols (w>>1)*16), O d-tiles dc*8+w (interleaved).
// Q and K staging use async global->LDS (bf16 -> bf16, no conversion).
// ---------------------------------------------------------------------------
__global__ __launch_bounds__(256)
void mha_flash_attn(const unsigned short* __restrict__ e,
                    unsigned short* __restrict__ o) {
  constexpr int LQ = 776, LKC = 136, LVT = 72, LS = 68, LP = 72;
  __shared__ __attribute__((aligned(16))) unsigned short Qs[32 * LQ];
  __shared__ __attribute__((aligned(16))) unsigned short Ks[64 * LKC];
  __shared__ __attribute__((aligned(16))) unsigned short Vt[128 * LVT];  // [d][key]
  __shared__ __attribute__((aligned(16))) float          Sls[32 * LS];
  __shared__ __attribute__((aligned(16))) unsigned short Pls[32 * LP];
  __shared__ float rowMax[32], rowSum[32], rowAlpha[32];

  const int t = threadIdx.x;
  const int w = t >> 5, lane = t & 31, ln = lane & 15, hi = lane >> 4;
  const int qb = blockIdx.x;   // 0..127
  const int b  = blockIdx.y;   // 0..3
  const int h  = blockIdx.z;   // 0..2

  const size_t eb = ((size_t)h * 4 + b) * 12288ull * 768ull;
  const unsigned short* Qg = e + eb + (size_t)(qb * 32) * 768;
  const unsigned short* Kg = e + eb + 4096ull * 768ull;
  const unsigned short* Vg = e + eb + 8192ull * 768ull;

  // ---- async-stage Q block: 32 rows x 1536 B (96 x 16B chunks per row)
  {
    const int r = t >> 3, sub = t & 7;           // 8 threads per row
    unsigned ldsrow = lds_off_of(&Qs[r * LQ]);
    unsigned grow = (unsigned)(r * 1536);
#pragma unroll
    for (int i = 0; i < 12; ++i) {
      unsigned cb = (unsigned)((sub + i * 8) * 16);
      async_ld_b128(ldsrow + cb, grow + cb, Qg);
    }
  }
  if (t < 32) { rowMax[t] = -3.0e38f; rowSum[t] = 0.0f; }
  wait_async0();
  __syncthreads();

  const int rt  = w & 1;
  const int ctk = w >> 1;

  v8f oacc[2][6] = {};

  for (int kc = 0; kc < 64; ++kc) {
    // ---- S = (Q K^T) tile, accumulate over d in 6 chunks of 128
    v8f sacc = {};
    for (int dc = 0; dc < 6; ++dc) {
      __syncthreads();
      // async-stage K chunk: 64 rows x 256 B (16 x 16B chunks per row)
      for (int c = t; c < 1024; c += 256) {
        int r = c >> 4, k16 = c & 15;
        async_ld_b128(lds_off_of(&Ks[r * LKC]) + (unsigned)(k16 * 16),
                      (unsigned)(((kc * 64 + r) * 768 + dc * 128) * 2 + k16 * 16),
                      Kg);
      }
      wait_async0();
      __syncthreads();
      const unsigned short* qrow = &Qs[(rt * 16 + ln) * LQ + dc * 128];
      const unsigned short* kcol = &Ks[(ctk * 16 + ln) * LKC];
#pragma unroll
      for (int kt = 0; kt < 4; ++kt) {
        u16x16 af = load_a_frag(qrow, kt * 32, hi);
        u16x16 bf = load_b_frag(kcol, kt * 32, hi);
        sacc = wmma_bf16(af, bf, sacc);
      }
    }
#pragma unroll
    for (int r = 0; r < 8; ++r) {
      int row = rt * 16 + r + 8 * hi;
      Sls[row * LS + ctk * 16 + ln] = sacc[r] * 0.0625f;
    }
    __syncthreads();

    // ---- online softmax, one thread per row
    if (t < 32) {
      float mOld = rowMax[t];
      float mNew = mOld;
      for (int j = 0; j < 64; ++j) mNew = fmaxf(mNew, Sls[t * LS + j]);
      float alpha = __expf(mOld - mNew);
      float sum = 0.0f;
      for (int j = 0; j < 64; ++j) {
        float p = __expf(Sls[t * LS + j] - mNew);
        Pls[t * LP + j] = f2bf(p);
        sum += p;
      }
      rowSum[t]   = rowSum[t] * alpha + sum;
      rowMax[t]   = mNew;
      rowAlpha[t] = alpha;
    }
    __syncthreads();

    // ---- rescale O accumulators
#pragma unroll
    for (int rt2 = 0; rt2 < 2; ++rt2) {
#pragma unroll
      for (int r = 0; r < 8; ++r) {
        float a = rowAlpha[rt2 * 16 + r + 8 * hi];
#pragma unroll
        for (int dc = 0; dc < 6; ++dc) oacc[rt2][dc][r] *= a;
      }
    }

    // ---- O += P V, d in 6 chunks of 128 (wave w uses local d-tile w)
    // V is transposed into LDS ([d][key]) -> manual staging path.
    for (int dc = 0; dc < 6; ++dc) {
      __syncthreads();
      for (int j = t; j < 64 * 128; j += 256) {
        int r = j >> 7, d = j & 127;
        Vt[d * LVT + r] = Vg[(size_t)(kc * 64 + r) * 768 + dc * 128 + d];
      }
      __syncthreads();
      const unsigned short* vcol = &Vt[(w * 16 + ln) * LVT];
#pragma unroll
      for (int rt2 = 0; rt2 < 2; ++rt2) {
        const unsigned short* prow = &Pls[(rt2 * 16 + ln) * LP];
#pragma unroll
        for (int ks = 0; ks < 2; ++ks) {
          u16x16 af = load_a_frag(prow, ks * 32, hi);
          u16x16 bf = load_b_frag(vcol, ks * 32, hi);
          oacc[rt2][dc] = wmma_bf16(af, bf, oacc[rt2][dc]);
        }
      }
    }
  }
  __syncthreads();

  // ---- normalize and store; heads concatenated along sequence
  unsigned short* Og = o + ((size_t)b * 12288 + (size_t)h * 4096 + qb * 32) * 768ull;
#pragma unroll
  for (int rt2 = 0; rt2 < 2; ++rt2) {
#pragma unroll
    for (int r = 0; r < 8; ++r) {
      int row = rt2 * 16 + r + 8 * hi;
      float linv = 1.0f / rowSum[row];
#pragma unroll
      for (int dc = 0; dc < 6; ++dc) {
        int col = dc * 128 + w * 16 + ln;
        Og[(size_t)row * 768 + col] = f2bf(oacc[rt2][dc][r] * linv);
      }
    }
  }
}

// ---------------------------------------------------------------------------
// Kernel C: out[row, c] = qkv[row, :] . W_out[c, :] + b_out[c] + x[row, c]
// rows = 49152, K = 768 (streamed), N = 256. 128x128 tiles.
// A operand (bf16) staged with async global->LDS; W_out needs f32->bf16.
// ---------------------------------------------------------------------------
__global__ __launch_bounds__(256)
void mha_out_proj(const unsigned short* __restrict__ o,
                  const float* __restrict__ W_out,
                  const float* __restrict__ b_out,
                  const float* __restrict__ x,
                  float* __restrict__ out) {
  constexpr int LD = 136;  // 128 + 8
  __shared__ __attribute__((aligned(16))) unsigned short As[128 * LD];
  __shared__ __attribute__((aligned(16))) unsigned short Bs[128 * LD];

  const int t = threadIdx.x;
  const int w = t >> 5, lane = t & 31, ln = lane & 15, hi = lane >> 4;
  const int rowBase = blockIdx.x * 128;
  const int cBase   = blockIdx.y * 128;

  v8f acc[8] = {};
  for (int kcb = 0; kcb < 6; ++kcb) {
    __syncthreads();
    // async-stage A chunk: 128 rows x 256 B (16 x 16B chunks per row)
    for (int c = t; c < 2048; c += 256) {
      int r = c >> 4, k16 = c & 15;
      async_ld_b128(lds_off_of(&As[r * LD]) + (unsigned)(k16 * 16),
                    (unsigned)(((unsigned)(rowBase + r) * 768u + (unsigned)(kcb * 128)) * 2u
                               + (unsigned)(k16 * 16)),
                    o);
    }
    // W_out chunk: f32 -> bf16 (VALU path)
    for (int j = t; j < 128 * 128; j += 256) {
      int r = j >> 7, k = j & 127;
      Bs[r * LD + k] = f2bf(W_out[(size_t)(cBase + r) * 768 + kcb * 128 + k]);
    }
    wait_async0();
    __syncthreads();
    const unsigned short* arow = &As[(w * 16 + ln) * LD];
#pragma unroll
    for (int kt = 0; kt < 4; ++kt) {
      u16x16 af = load_a_frag(arow, kt * 32, hi);
#pragma unroll
      for (int ct = 0; ct < 8; ++ct) {
        u16x16 bf = load_b_frag(&Bs[(ct * 16 + ln) * LD], kt * 32, hi);
        acc[ct] = wmma_bf16(af, bf, acc[ct]);
      }
    }
  }

#pragma unroll
  for (int ct = 0; ct < 8; ++ct) {
#pragma unroll
    for (int r = 0; r < 8; ++r) {
      int row = rowBase + w * 16 + r + 8 * hi;
      int col = cBase + ct * 16 + ln;
      size_t idx = (size_t)row * 256 + col;
      out[idx] = acc[ct][r] + b_out[col] + x[idx];
    }
  }
}

// ---------------------------------------------------------------------------
extern "C" void kernel_launch(void* const* d_in, const int* in_sizes, int n_in,
                              void* d_out, int out_size, void* d_ws, size_t ws_size,
                              hipStream_t stream) {
  (void)in_sizes; (void)n_in; (void)out_size; (void)ws_size;
  const float* x       = (const float*)d_in[0];  // [4,12288,256]
  const float* W_embed = (const float*)d_in[1];  // [3,768,256]
  const float* b_embed = (const float*)d_in[2];  // [3,768]
  const float* W_out   = (const float*)d_in[3];  // [256,768]
  const float* b_out   = (const float*)d_in[4];  // [256]
  float* out = (float*)d_out;                    // [4,12288,256]

  unsigned short* e = (unsigned short*)d_ws;                    // [3][49152][768] bf16
  unsigned short* o = e + (size_t)3 * 49152 * 768;              // [49152][768] bf16

  mha_embed_gemm<<<dim3(384, 6, 3), 256, 0, stream>>>(x, W_embed, b_embed, e);
  mha_flash_attn<<<dim3(128, 4, 3), 256, 0, stream>>>(e, o);
  mha_out_proj <<<dim3(384, 2, 1), 256, 0, stream>>>(o, W_out, b_out, x, out);
}
